// RNN_68513318306065
// MI455X (gfx1250) — matrix operand order repaired and checked by reading, not compiled, not run on hardware
//
#include <hip/hip_runtime.h>
#include <cmath>

// ---------------- types ----------------
typedef __attribute__((ext_vector_type(16))) __bf16   v16bf;
typedef __attribute__((ext_vector_type(8)))  float    v8f;
typedef __attribute__((ext_vector_type(4)))  unsigned v4u;
typedef __attribute__((ext_vector_type(2)))  unsigned v2u;
typedef __attribute__((ext_vector_type(4)))  float    v4f;
typedef __attribute__((ext_vector_type(4)))  unsigned u32x4;
typedef __attribute__((ext_vector_type(8)))  int      i32x8;
typedef __attribute__((ext_vector_type(4)))  int      i32x4;

// ---------------- problem constants ----------------
#define B_   128
#define T_   1024
#define I_   256
#define H_   512
#define C_   10
#define NWG  16
#define NTHR 256

// ---------------- workspace layout (bf16 element offsets) ----------------
#define OFF_WIH1 0          // 512*256  = 131072
#define OFF_WHH1 131072     // 512*512  = 262144
#define OFF_WIH2 393216     // 262144
#define OFF_WHH2 655360     // 262144
#define OFF_H1   917504     // 2 * 128*512 = 131072 (double buffered)
#define OFF_H2   1048576    // 131072
#define WS_ELEMS 1179648
#define SYNC_BYTE (WS_ELEMS * 2)   // unsigned counter after bf16 region

// ---------------- LDS layout (bf16 element offsets, padded rows) ----------
#define SM_IH1 0            // 32 rows * (256+8) = 8448
#define SM_HH1 8448         // 32 rows * (512+8) = 16640
#define SM_IH2 25088        // 16640
#define SM_HH2 41728        // 16640
#define SM_TOT 58368        // 116736 bytes of the 320KB WGP LDS

#if __has_builtin(__builtin_amdgcn_tensor_load_to_lds)
#define HAVE_TDM 1
#else
#define HAVE_TDM 0
#endif

union F16 {
    v16bf  v;
    __bf16 h[16];
    v4u    q[2];
};

__device__ __forceinline__ v8f zero8() { v8f z = {}; return z; }

// Native gfx1250 transcendental tanh; v_nop covers the 1-op TRANS hazard
// (result may not be consumed by the immediately following instruction).
__device__ __forceinline__ float fast_tanh(float v) {
    float r;
    asm volatile("v_tanh_f32 %0, %1\n\tv_nop" : "=v"(r) : "v"(v));
    return r;
}

__device__ __forceinline__ v8f wmma_bf16(v16bf a, v16bf b, v8f c) {
    // D = A(16x32 bf16) x B(32x16 bf16) + C(f32)
    return __builtin_amdgcn_wmma_f32_16x16x32_bf16(
        /*neg_a=*/false, a, /*neg_b=*/false, b,
        /*c_mod=*/(short)0, c, /*reuse_a=*/false, /*reuse_b=*/false);
}

// A-fragment (16x32, bf16, row-major source). p already offset to
// row*stride + k0 + 8*laneHi.  elems[0..7] = K..K+7, elems[8..15] = K+16..K+23.
__device__ __forceinline__ v16bf load_a_h(const __bf16* p) {
    F16 u;
    u.q[0] = *(const v4u*)(p);
    u.q[1] = *(const v4u*)(p + 16);
    return u.v;
}

// Same A layout, but source is fp32 (x): convert on the fly.
__device__ __forceinline__ v16bf load_a_f32(const float* p) {
    F16 u;
    v4f f0 = *(const v4f*)(p);
    v4f f1 = *(const v4f*)(p + 4);
    v4f f2 = *(const v4f*)(p + 16);
    v4f f3 = *(const v4f*)(p + 20);
#pragma unroll
    for (int j = 0; j < 4; ++j) {
        u.h[j]      = (__bf16)f0[j];
        u.h[4 + j]  = (__bf16)f1[j];
        u.h[8 + j]  = (__bf16)f2[j];
        u.h[12 + j] = (__bf16)f3[j];
    }
    return u.v;
}

// B-fragment (32x16, bf16) from LDS.  p already offset to
// n*strideP + k0 + 8*laneHi ; elems[0..7] = K..K+7, elems[8..15] = K+16..K+23.
__device__ __forceinline__ v16bf load_b(const __bf16* p) {
    F16 u;
    u.q[0] = *(const v4u*)(p);
    u.q[1] = *(const v4u*)(p + 16);
    return u.v;
}

#if HAVE_TDM
// TDM: DMA one [32 x K] bf16 weight slice into LDS, padding each row of K
// halves with 8 halves (4 DWORDs) to reproduce the bank-conflict-free layout.
//   K=256 -> pad_interval code 6 (128 DWORDs); K=512 -> code 7 (256 DWORDs);
//   pad_amount code 3 = 4 DWORDs.
// This toolchain declares the 6-arg builtin:
//   (u32x4 g0, i32x8 g1, i32x4 g2, i32x4 g3, i32x8 g4, i32 cpol)
__device__ __forceinline__ void tdm_load_w(unsigned lds_byte_off, const __bf16* gsrc,
                                           unsigned K, unsigned intervalCode) {
    unsigned long long ga = (unsigned long long)(size_t)gsrc;
    u32x4 g0;
    g0[0] = 1u;                                   // count=1, user descriptor
    g0[1] = lds_byte_off;                         // lds_addr (bytes)
    g0[2] = (unsigned)ga;                         // global_addr[31:0]
    g0[3] = (unsigned)((ga >> 32) & 0x1FFFFFFu)   // global_addr[56:32]
          | (2u << 30);                           // type = 2 ("image")
    i32x8 g1;
    g1[0] = (int)((1u << 16)                      // data_size = 1 (2 bytes)
                | (1u << 20)                      // pad_enable
                | (intervalCode << 22)            // pad_interval
                | (3u << 25));                    // pad_amount = 4 DWORDs
    g1[1] = (int)((K & 0xFFFFu) << 16);           // tensor_dim0[15:0] @ bits 63:48
    g1[2] = (int)(32u << 16);                     // tensor_dim1 = 32 @ bits 95:80
    g1[3] = (int)(K << 16);                       // tile_dim0 = K @ bits 127:112
    g1[4] = (int)32u;                             // tile_dim1 = 32 @ bits 143:128
    g1[5] = (int)K;                               // tensor_dim0_stride = K
    g1[6] = 0;
    g1[7] = 0;
    i32x4 z4 = {0, 0, 0, 0};                      // groups 2/3 unused (2D tensor)
    i32x8 z8 = {0, 0, 0, 0, 0, 0, 0, 0};
    __builtin_amdgcn_tensor_load_to_lds(g0, g1, z4, z4, z8, 0);
}

__device__ __forceinline__ void wait_tensor0() {
#if __has_builtin(__builtin_amdgcn_s_wait_tensorcnt)
    __builtin_amdgcn_s_wait_tensorcnt((short)0);
#else
    asm volatile("s_wait_tensorcnt 0x0" ::: "memory");
#endif
}
#endif  // HAVE_TDM

// Fallback / reference path: stage a [32 x K] weight slice with vector copies.
__device__ __forceinline__ void load_w(__bf16* sm, const __bf16* g,
                                       int colbase, int K, int strideP) {
    const int kc = K >> 2;            // 4-half (8B) chunks per row
    const int chunks = 32 * kc;
    for (int c = threadIdx.x; c < chunks; c += NTHR) {
        int n  = c / kc;
        int k4 = (c % kc) * 4;
        *(v2u*)(sm + n * strideP + k4) =
            *(const v2u*)(g + (size_t)(colbase + n) * K + k4);
    }
}

// Device-wide barrier over NWG co-resident workgroups (monotonic counter).
__device__ __forceinline__ void grid_sync(unsigned* cnt, unsigned target) {
    __syncthreads();
    if (threadIdx.x == 0) {
        __threadfence();   // make this WG's global stores visible
        __hip_atomic_fetch_add(cnt, 1u, __ATOMIC_RELEASE, __HIP_MEMORY_SCOPE_AGENT);
        while (__hip_atomic_load(cnt, __ATOMIC_ACQUIRE, __HIP_MEMORY_SCOPE_AGENT) < target)
            __builtin_amdgcn_s_sleep(2);
    }
    __syncthreads();
    __threadfence();       // acquire side for all lanes
}

// ---------------- prep: fp32->bf16 weights, zero h state, reset sync -------
__global__ void __launch_bounds__(NTHR)
rnn_prep_kernel(const float* __restrict__ wih1, const float* __restrict__ whh1,
                const float* __restrict__ wih2, const float* __restrict__ whh2,
                __bf16* __restrict__ ws, unsigned* __restrict__ sync) {
    unsigned i      = blockIdx.x * blockDim.x + threadIdx.x;
    unsigned stride = gridDim.x * blockDim.x;
    for (unsigned k = i; k < 262144u; k += stride) {
        if (k < 131072u) ws[OFF_WIH1 + k] = (__bf16)wih1[k];
        ws[OFF_WHH1 + k] = (__bf16)whh1[k];
        ws[OFF_WIH2 + k] = (__bf16)wih2[k];
        ws[OFF_WHH2 + k] = (__bf16)whh2[k];
        ws[OFF_H1 + k]   = (__bf16)0.0f;   // zeros h1[2] then h2[2] (contiguous)
    }
    if (i == 0) *sync = 0u;
}

// ---------------- persistent RNN kernel: 16 WGs x 256 threads --------------
__global__ void __launch_bounds__(NTHR)
rnn_persistent_kernel(const float* __restrict__ x,
                      const float* __restrict__ b_ih1, const float* __restrict__ b_hh1,
                      const float* __restrict__ b_ih2, const float* __restrict__ b_hh2,
                      const float* __restrict__ w_out, const float* __restrict__ b_out,
                      __bf16* __restrict__ ws, unsigned* __restrict__ sync,
                      float* __restrict__ out) {
    const int tid     = threadIdx.x;
    const int g       = blockIdx.x;        // 0..15, owns 32 output columns
    const int wv      = tid >> 5;          // wave 0..7, owns 16 batch rows
    const int lane    = tid & 31;
    const int laneN   = lane & 15;
    const int laneHi  = lane >> 4;
    const int rb      = wv * 16;
    const int colbase = g * 32;
    const int rowA    = rb + laneN;        // A-matrix row carried by this lane

    __shared__ __attribute__((aligned(16))) __bf16 smem[SM_TOT];

    // Stage this WG's weight slices into LDS once; reused for all 1024 steps.
#if HAVE_TDM
    if (wv == 0) {
        tdm_load_w((unsigned)(size_t)&smem[SM_IH1], ws + OFF_WIH1 + (size_t)colbase * I_, I_, 6u);
        tdm_load_w((unsigned)(size_t)&smem[SM_HH1], ws + OFF_WHH1 + (size_t)colbase * H_, H_, 7u);
        tdm_load_w((unsigned)(size_t)&smem[SM_IH2], ws + OFF_WIH2 + (size_t)colbase * H_, H_, 7u);
        tdm_load_w((unsigned)(size_t)&smem[SM_HH2], ws + OFF_WHH2 + (size_t)colbase * H_, H_, 7u);
        wait_tensor0();
    }
#else
    load_w(smem + SM_IH1, ws + OFF_WIH1, colbase, I_, 264);
    load_w(smem + SM_HH1, ws + OFF_WHH1, colbase, H_, 520);
    load_w(smem + SM_IH2, ws + OFF_WIH2, colbase, H_, 520);
    load_w(smem + SM_HH2, ws + OFF_WHH2, colbase, H_, 520);
#endif
    __syncthreads();

    // Per-lane bias sums (C/D layout: every acc element has N = laneN).
    const float bs1_0 = b_ih1[colbase + laneN]      + b_hh1[colbase + laneN];
    const float bs1_1 = b_ih1[colbase + 16 + laneN] + b_hh1[colbase + 16 + laneN];
    const float bs2_0 = b_ih2[colbase + laneN]      + b_hh2[colbase + laneN];
    const float bs2_1 = b_ih2[colbase + 16 + laneN] + b_hh2[colbase + 16 + laneN];

    __bf16* hp1 = ws + OFF_H1;                 // h1 ping-pong (bf16 [B,H])
    __bf16* hq1 = ws + OFF_H1 + B_ * H_;
    __bf16* hp2 = ws + OFF_H2;                 // h2 ping-pong
    __bf16* hq2 = ws + OFF_H2 + B_ * H_;

    unsigned ph = 1;

    for (int t = 0; t < T_; ++t) {
        // ================= cell 1: tanh(x_t W_ih1^T + h1 W_hh1^T + b) ======
        v8f acc0 = zero8(), acc1 = zero8();

        const float* xrow = x + ((size_t)rowA * T_ + t) * I_;
        if (t + 1 < T_) {
            // prefetch next timestep's x row (global_prefetch_b8)
            __builtin_prefetch(xrow + I_, 0, 0);
            __builtin_prefetch(xrow + I_ + 128, 0, 0);
        }
#pragma unroll 2
        for (int kk = 0; kk < 8; ++kk) {       // K over I_ = 256
            v16bf a  = load_a_f32(xrow + kk * 32 + laneHi * 8);
            v16bf b0 = load_b(smem + SM_IH1 + laneN * 264        + kk * 32 + laneHi * 8);
            v16bf b1 = load_b(smem + SM_IH1 + (16 + laneN) * 264 + kk * 32 + laneHi * 8);
            acc0 = wmma_bf16(a, b0, acc0);
            acc1 = wmma_bf16(a, b1, acc1);
        }
        const __bf16* h1row = hp1 + (size_t)rowA * H_;
#pragma unroll 2
        for (int kk = 0; kk < 16; ++kk) {      // K over H_ = 512
            v16bf a  = load_a_h(h1row + kk * 32 + laneHi * 8);
            v16bf b0 = load_b(smem + SM_HH1 + laneN * 520        + kk * 32 + laneHi * 8);
            v16bf b1 = load_b(smem + SM_HH1 + (16 + laneN) * 520 + kk * 32 + laneHi * 8);
            acc0 = wmma_bf16(a, b0, acc0);
            acc1 = wmma_bf16(a, b1, acc1);
        }
#pragma unroll
        for (int e = 0; e < 8; ++e) {          // acc elem e -> row M = 8*laneHi+e
            int row = rb + laneHi * 8 + e;
            hq1[(size_t)row * H_ + colbase + laneN]      = (__bf16)fast_tanh(acc0[e] + bs1_0);
            hq1[(size_t)row * H_ + colbase + 16 + laneN] = (__bf16)fast_tanh(acc1[e] + bs1_1);
        }
        grid_sync(sync, NWG * ph); ++ph;       // h1(t) fully visible

        // ================= cell 2: tanh(h1 W_ih2^T + h2 W_hh2^T + b) =======
        acc0 = zero8(); acc1 = zero8();
        const __bf16* h1nrow = hq1 + (size_t)rowA * H_;
        const __bf16* h2row  = hp2 + (size_t)rowA * H_;
#pragma unroll 2
        for (int kk = 0; kk < 16; ++kk) {
            v16bf a  = load_a_h(h1nrow + kk * 32 + laneHi * 8);
            v16bf b0 = load_b(smem + SM_IH2 + laneN * 520        + kk * 32 + laneHi * 8);
            v16bf b1 = load_b(smem + SM_IH2 + (16 + laneN) * 520 + kk * 32 + laneHi * 8);
            acc0 = wmma_bf16(a, b0, acc0);
            acc1 = wmma_bf16(a, b1, acc1);
        }
#pragma unroll 2
        for (int kk = 0; kk < 16; ++kk) {
            v16bf a  = load_a_h(h2row + kk * 32 + laneHi * 8);
            v16bf b0 = load_b(smem + SM_HH2 + laneN * 520        + kk * 32 + laneHi * 8);
            v16bf b1 = load_b(smem + SM_HH2 + (16 + laneN) * 520 + kk * 32 + laneHi * 8);
            acc0 = wmma_bf16(a, b0, acc0);
            acc1 = wmma_bf16(a, b1, acc1);
        }
#pragma unroll
        for (int e = 0; e < 8; ++e) {
            int row = rb + laneHi * 8 + e;
            hq2[(size_t)row * H_ + colbase + laneN]      = (__bf16)fast_tanh(acc0[e] + bs2_0);
            hq2[(size_t)row * H_ + colbase + 16 + laneN] = (__bf16)fast_tanh(acc1[e] + bs2_1);
        }
        grid_sync(sync, NWG * ph); ++ph;       // h2(t) fully visible

        // swap ping-pong buffers
        __bf16* tmp;
        tmp = hp1; hp1 = hq1; hq1 = tmp;
        tmp = hp2; hp2 = hq2; hq2 = tmp;
    }

    // ================= output: out = h2_final @ w_out^T + b_out ============
    const __bf16* hf = hp2;                    // final h2 after last swap
    for (int it = tid; it < 80; it += NTHR) {  // 16 WGs * 80 = 1280 = 128*10
        int item = g * 80 + it;
        int b = item / C_, c = item % C_;
        float s = b_out[c];
        for (int h = 0; h < H_; ++h)
            s += (float)hf[(size_t)b * H_ + h] * w_out[c * H_ + h];
        out[item] = s;
    }
}

// ---------------- host launcher -------------------------------------------
extern "C" void kernel_launch(void* const* d_in, const int* in_sizes, int n_in,
                              void* d_out, int out_size, void* d_ws, size_t ws_size,
                              hipStream_t stream) {
    const float* x     = (const float*)d_in[0];
    const float* w_ih1 = (const float*)d_in[1];
    const float* w_hh1 = (const float*)d_in[2];
    const float* b_ih1 = (const float*)d_in[3];
    const float* b_hh1 = (const float*)d_in[4];
    const float* w_ih2 = (const float*)d_in[5];
    const float* w_hh2 = (const float*)d_in[6];
    const float* b_ih2 = (const float*)d_in[7];
    const float* b_hh2 = (const float*)d_in[8];
    const float* w_out = (const float*)d_in[9];
    const float* b_out = (const float*)d_in[10];
    float*       out   = (float*)d_out;

    __bf16*   ws   = (__bf16*)d_ws;
    unsigned* sync = (unsigned*)((char*)d_ws + SYNC_BYTE);

    rnn_prep_kernel<<<1024, NTHR, 0, stream>>>(w_ih1, w_hh1, w_ih2, w_hh2, ws, sync);
    rnn_persistent_kernel<<<NWG, NTHR, 0, stream>>>(
        x, b_ih1, b_hh1, b_ih2, b_hh2, w_out, b_out, ws, sync, out);
}
